// Model_33054068310765
// MI455X (gfx1250) — matrix operand order repaired and checked by reading, not compiled, not run on hardware
//
#include <hip/hip_runtime.h>
#include <hip/hip_bf16.h>

typedef _Float16 v16h __attribute__((ext_vector_type(16)));
typedef _Float16 h8   __attribute__((ext_vector_type(8)));
typedef float    v8f  __attribute__((ext_vector_type(8)));

#define NB   64      // batch
#define DM   512     // d_model
#define HHD  768     // n_heads * d_k
#define NHEADS 6
#define DKH  128     // d_k
#define LE   168     // encoder input length
#define SEQE 222     // encoder token count (168+42+10+2)
#define LDQ  168     // decoder length
#define SKV  390     // decoder kv length (222+168)
#define MQE  224     // padded enc queries (mult of 16)
#define NKE  256     // padded enc keys    (mult of 128)
#define MQD  176     // padded dec queries (mult of 16)
#define NKD  512     // padded dec keys    (mult of 128)
#define ZBUF_HALFS 2048

// ---------------------------------------------------------------------------
// Batched WMMA GEMM: C[M,N](f32) = A[M,K](f16) * Bt[N,K](f16)^T + bias[N]
// One wave computes a 16x32 tile (2 independent accumulators, shared A frag);
// 4 waves/block cover 128 columns.
// grid = (N/128, M/16, batches);  batch z -> (b = z/nh, h = z%nh)
// Out-of-range rows (A or Bt) are redirected to a zero-filled scratch region
// so no per-iteration masking is needed and EXEC stays all-ones for WMMA.
// ---------------------------------------------------------------------------
__global__ void __launch_bounds__(128)
k_wmma_gemm(const _Float16* __restrict__ A, int lda, long long sAb, long long sAh, int validM,
            const _Float16* __restrict__ Bt, int ldb, long long sBb, long long sBh, int validN,
            float* __restrict__ C, int ldc, long long sCb, long long sCh,
            const float* __restrict__ bias, int K, int nh,
            const _Float16* __restrict__ zbuf)
{
    const int lane = threadIdx.x & 31;
    const int wave = threadIdx.x >> 5;
    const int n0 = blockIdx.x * 128 + wave * 32;
    const int m0 = blockIdx.y * 16;
    const int z  = blockIdx.z;
    const int b  = z / nh, hh = z - b * nh;
    const int l16 = lane & 15, hi = lane >> 4;

    const int am  = m0 + l16;
    const int bn0 = n0 + l16;
    const int bn1 = n0 + 16 + l16;

    // A fragment: row m, K chunks [kb..kb+7] and [kb+16..kb+23], kb = hi*8
    const _Float16* __restrict__ arow = (am < validM)
        ? (A + (long long)b * sAb + (long long)hh * sAh + (long long)am * lda + hi * 8)
        : zbuf;
    // B fragments: Bt row n, K chunks [kb..kb+7] and [kb+16..kb+23], kb = hi*8
    const _Float16* __restrict__ Bb = Bt + (long long)b * sBb + (long long)hh * sBh;
    const _Float16* __restrict__ brow0 = (bn0 < validN) ? (Bb + (long long)bn0 * ldb + hi * 8) : zbuf;
    const _Float16* __restrict__ brow1 = (bn1 < validN) ? (Bb + (long long)bn1 * ldb + hi * 8) : zbuf;

    v8f acc0 = {0.f, 0.f, 0.f, 0.f, 0.f, 0.f, 0.f, 0.f};
    v8f acc1 = {0.f, 0.f, 0.f, 0.f, 0.f, 0.f, 0.f, 0.f};

    for (int k0 = 0; k0 < K; k0 += 32) {
        h8 a0 = *(const h8*)(arow + k0);
        h8 a1 = *(const h8*)(arow + k0 + 16);
        h8 p0 = *(const h8*)(brow0 + k0);
        h8 p1 = *(const h8*)(brow0 + k0 + 16);
        h8 q0 = *(const h8*)(brow1 + k0);
        h8 q1 = *(const h8*)(brow1 + k0 + 16);
        __builtin_prefetch(arow + k0 + 128, 0, 1);    // global_prefetch_b8
        __builtin_prefetch(brow0 + k0 + 128, 0, 1);
        __builtin_prefetch(brow1 + k0 + 128, 0, 1);
        v16h af  = __builtin_shufflevector(a0, a1, 0,1,2,3,4,5,6,7,8,9,10,11,12,13,14,15);
        v16h bf0 = __builtin_shufflevector(p0, p1, 0,1,2,3,4,5,6,7,8,9,10,11,12,13,14,15);
        v16h bf1 = __builtin_shufflevector(q0, q1, 0,1,2,3,4,5,6,7,8,9,10,11,12,13,14,15);
        // two independent accumulator chains -> co-exec slots cover the
        // WMMA->WMMA data-hazard latency without v_nops
        acc0 = __builtin_amdgcn_wmma_f32_16x16x32_f16(false, af, false, bf0,
                                                      (short)0, acc0, false, false);
        acc1 = __builtin_amdgcn_wmma_f32_16x16x32_f16(false, af, false, bf1,
                                                      (short)0, acc1, false, false);
    }

    const float bb0 = bias ? bias[n0 + l16] : 0.f;
    const float bb1 = bias ? bias[n0 + 16 + l16] : 0.f;
    float* __restrict__ cbase = C + (long long)b * sCb + (long long)hh * sCh
                                  + (long long)(m0 + hi * 8) * ldc;
    float* __restrict__ cp0 = cbase + (n0 + l16);
    float* __restrict__ cp1 = cbase + (n0 + 16 + l16);
#pragma unroll
    for (int r = 0; r < 8; ++r) {
        cp0[(long long)r * ldc] = acc0[r] + bb0;
        cp1[(long long)r * ldc] = acc1[r] + bb1;
    }
}

__global__ void k_zero16(_Float16* __restrict__ p, int n)
{
    int i = blockIdx.x * 256 + threadIdx.x;
    if (i < n) p[i] = (_Float16)0.f;
}

// ---------------------------------------------------------------------------
// Weight transpose + f32->f16:  W[K,N] (row-major) -> Wt[N,K] f16
// ---------------------------------------------------------------------------
__global__ void k_wT(const float* __restrict__ W, _Float16* __restrict__ Wt, int K, int N)
{
    long long i = (long long)blockIdx.x * 256 + threadIdx.x;
    if (i >= (long long)K * N) return;
    int n = (int)(i / K), k = (int)(i - (long long)n * K);
    Wt[i] = (_Float16)W[(long long)k * N + n];
}

__global__ void k_h(const float* __restrict__ s, _Float16* __restrict__ d, long long n)
{
    long long i = (long long)blockIdx.x * 256 + threadIdx.x;
    if (i < n) d[i] = (_Float16)s[i];
}

__global__ void k_gelu(const float* __restrict__ s, _Float16* __restrict__ d, long long n)
{
    long long i = (long long)blockIdx.x * 256 + threadIdx.x;
    if (i >= n) return;
    float x = s[i];
    d[i] = (_Float16)(0.5f * x * (1.f + erff(x * 0.70710678118654752f)));
}

// ---------------------------------------------------------------------------
// Embedding: circular conv1d(k=3) over 7ch + sinusoidal PE + time-feature linear
// block = 512 threads (one per d), one block per (b,l)
// ---------------------------------------------------------------------------
__global__ void __launch_bounds__(512)
k_embed(const float* __restrict__ x, const float* __restrict__ t,
        const float* __restrict__ cw, const float* __restrict__ tw,
        const float* __restrict__ tb, float* __restrict__ o32,
        _Float16* __restrict__ o16, int L)
{
    __shared__ float xs[21];
    __shared__ float ts[4];
    const int row = blockIdx.x;
    const int b = row / L, l = row - b * L;
    const int d = threadIdx.x;
    if (d < 21) {
        int j = d / 7, c = d - j * 7;
        int ll = l + j - 1; if (ll < 0) ll += L; if (ll >= L) ll -= L;
        xs[j * 7 + c] = x[((long long)b * L + ll) * 7 + c];
    }
    if (d < 4) ts[d] = t[((long long)b * L + l) * 4 + d];
    __syncthreads();
    float s = tb[d];
#pragma unroll
    for (int c = 0; c < 7; ++c)
#pragma unroll
        for (int j = 0; j < 3; ++j)
            s += xs[j * 7 + c] * cw[(d * 7 + c) * 3 + j];
#pragma unroll
    for (int f = 0; f < 4; ++f) s += ts[f] * tw[f * 512 + d];
    const int pair = d >> 1;
    float ang = (float)l * __expf(-(float)(2 * pair) * (9.210340371976184f / 512.f));
    s += (d & 1) ? __cosf(ang) : __sinf(ang);
    o32[(long long)row * 512 + d] = s;
    o16[(long long)row * 512 + d] = (_Float16)s;
}

// ---------------------------------------------------------------------------
// Bottleneck strided conv (k=4, s=4, 128->128ch) + BatchNorm(eval) + ELU
// one block per (b, lo), 128 threads (one per out channel)
// ---------------------------------------------------------------------------
__global__ void __launch_bounds__(128)
k_conv_bn_elu(const float* __restrict__ in, long long in_bstride,
              const float* __restrict__ w, const float* __restrict__ bias,
              const float* __restrict__ bng, const float* __restrict__ bnb,
              const float* __restrict__ bnm, const float* __restrict__ bnv,
              float* __restrict__ out, long long out_bstride, int out_off, int Lout)
{
    __shared__ float is[512];
    const int r = blockIdx.x;
    const int b = r / Lout, lo = r - b * Lout;
    const int co = threadIdx.x;
    const float* ip = in + (long long)b * in_bstride + (long long)lo * 4 * 128;
    for (int k = co; k < 512; k += 128) is[k] = ip[k];
    __syncthreads();
    float s = bias[co];
    for (int j = 0; j < 4; ++j)
        for (int ci = 0; ci < 128; ++ci)
            s += is[j * 128 + ci] * w[(co * 128 + ci) * 4 + j];
    s = (s - bnm[co]) * rsqrtf(bnv[co] + 1e-5f) * bng[co] + bnb[co];
    s = (s > 0.f) ? s : (__expf(s) - 1.f);                      // ELU
    out[(long long)b * out_bstride + (long long)(out_off + lo) * 128 + co] = s;
}

// ---------------------------------------------------------------------------
// LayerNorm kernels (row length 512, block 256)
// ---------------------------------------------------------------------------
__device__ __forceinline__ float blk_sum(float v, float* sh)
{
    int t = threadIdx.x;
    sh[t] = v; __syncthreads();
    for (int s = 128; s > 0; s >>= 1) { if (t < s) sh[t] += sh[t + s]; __syncthreads(); }
    float r = sh[0]; __syncthreads();
    return r;
}

__global__ void __launch_bounds__(256)
k_ln_res(const float* __restrict__ P, const float* __restrict__ R,
         const float* __restrict__ g, const float* __restrict__ be,
         float* __restrict__ o32, _Float16* __restrict__ o16, float eps)
{
    __shared__ float sh[256];
    const long long row = blockIdx.x;
    const int t = threadIdx.x;
    float x0 = P[row * 512 + t]       + R[row * 512 + t];
    float x1 = P[row * 512 + t + 256] + R[row * 512 + t + 256];
    float mean = blk_sum(x0 + x1, sh) * (1.f / 512.f);
    float d0 = x0 - mean, d1 = x1 - mean;
    float var = blk_sum(d0 * d0 + d1 * d1, sh) * (1.f / 512.f);
    float inv = rsqrtf(var + eps);
    float y0 = d0 * inv * g[t] + be[t];
    float y1 = d1 * inv * g[t + 256] + be[t + 256];
    o32[row * 512 + t] = y0;           o32[row * 512 + t + 256] = y1;
    o16[row * 512 + t] = (_Float16)y0; o16[row * 512 + t + 256] = (_Float16)y1;
}

__global__ void __launch_bounds__(256)
k_concat_ln(const float* __restrict__ E, const float* __restrict__ U,
            const float* __restrict__ g, const float* __restrict__ be,
            float* __restrict__ o32, _Float16* __restrict__ o16)
{
    __shared__ float sh[256];
    const int row = blockIdx.x;
    const int b = row / SEQE, l = row - b * SEQE;
    const float* src = (l < LE) ? (E + ((long long)b * LE + l) * 512)
                                : (U + ((long long)b * 54 + (l - LE)) * 512);
    const int t = threadIdx.x;
    float x0 = src[t], x1 = src[t + 256];
    float mean = blk_sum(x0 + x1, sh) * (1.f / 512.f);
    float d0 = x0 - mean, d1 = x1 - mean;
    float var = blk_sum(d0 * d0 + d1 * d1, sh) * (1.f / 512.f);
    float inv = rsqrtf(var + 1e-5f);
    float y0 = d0 * inv * g[t] + be[t];
    float y1 = d1 * inv * g[t + 256] + be[t + 256];
    o32[(long long)row * 512 + t] = y0;           o32[(long long)row * 512 + t + 256] = y1;
    o16[(long long)row * 512 + t] = (_Float16)y0; o16[(long long)row * 512 + t + 256] = (_Float16)y1;
}

// ---------------------------------------------------------------------------
// Pyraformer PAM mask (sizes 168/42/10/2, window 4, inner 3)
// ---------------------------------------------------------------------------
__device__ __forceinline__ bool pam_allowed(int i, int j)
{
    const int st[4] = {0, 168, 210, 220};
    const int sz[4] = {168, 42, 10, 2};
    int li = (i < 168) ? 0 : (i < 210) ? 1 : (i < 220) ? 2 : 3;
    int lj = (j < 168) ? 0 : (j < 210) ? 1 : (j < 220) ? 2 : 3;
    if (li == lj) return (j >= i - 1) && (j <= i + 1);
    if (li == lj + 1) {
        int p = i - st[li];
        int lo = st[lj] + p * 4;
        int hi = (p == sz[li] - 1) ? st[li] : (st[lj] + (p + 1) * 4);
        return (j >= lo) && (j < hi);
    }
    if (lj == li + 1) {
        int p = j - st[lj];
        int lo = st[li] + p * 4;
        int hi = (p == sz[lj] - 1) ? st[lj] : (st[li] + (p + 1) * 4);
        return (i >= lo) && (i < hi);
    }
    return false;
}

// masktype 0: encoder PAM; 1: decoder causal (k <= 222+q). Writes f16 probs (zero-padded).
__global__ void __launch_bounds__(256)
k_softmax(const float* __restrict__ S, _Float16* __restrict__ P,
          int Lq, int Lqp, int Lk, int Lkp, float scale, int masktype)
{
    __shared__ float sh[256];
    const int q = blockIdx.x, z = blockIdx.y, t = threadIdx.x;
    const float* row = S + ((long long)z * Lqp + q) * Lkp;
    _Float16* prow = P + ((long long)z * Lqp + q) * Lkp;
    if (q >= Lq) { for (int k = t; k < Lkp; k += 256) prow[k] = (_Float16)0.f; return; }
    float mx = -1e30f;
    for (int k = t; k < Lkp; k += 256) {
        bool ok = (k < Lk) && (masktype ? (k <= 222 + q) : pam_allowed(q, k));
        if (ok) mx = fmaxf(mx, row[k] * scale);
    }
    sh[t] = mx; __syncthreads();
    for (int s = 128; s > 0; s >>= 1) { if (t < s) sh[t] = fmaxf(sh[t], sh[t + s]); __syncthreads(); }
    mx = sh[0]; __syncthreads();
    float sum = 0.f;
    for (int k = t; k < Lkp; k += 256) {
        bool ok = (k < Lk) && (masktype ? (k <= 222 + q) : pam_allowed(q, k));
        float e = ok ? __expf(row[k] * scale - mx) : 0.f;
        prow[k] = (_Float16)e;
        sum += e;
    }
    sum = blk_sum(sum, sh);
    float inv = 1.f / sum;
    for (int k = t; k < Lkp; k += 256) prow[k] = (_Float16)((float)prow[k] * inv);
}

// V [B*Lk, 768] f32 -> Vt [B, 6, 128, Lkp] f16 (zero-padded keys)
__global__ void k_vt(const float* __restrict__ V, _Float16* __restrict__ Vt,
                     int Lk, int Lkp, long long n)
{
    long long i = (long long)blockIdx.x * 256 + threadIdx.x;
    if (i >= n) return;
    int lk = (int)(i % Lkp); long long r = i / Lkp;
    int d = (int)(r % 128); r /= 128;
    int h = (int)(r % 6);   int b = (int)(r / 6);
    Vt[i] = (_Float16)((lk < Lk) ? V[((long long)b * Lk + lk) * 768 + h * 128 + d] : 0.f);
}

// AV [B,6,Lqp,128] f32 -> merged [B*Lq, 768] f16
__global__ void k_merge(const float* __restrict__ AV, _Float16* __restrict__ O,
                        int Lq, int Lqp, long long n)
{
    long long i = (long long)blockIdx.x * 256 + threadIdx.x;
    if (i >= n) return;
    int nn = (int)(i % 768); long long r = i / 768;
    int q = (int)(r % Lq);  int b = (int)(r / Lq);
    int h = nn >> 7, d = nn & 127;
    O[i] = (_Float16)AV[((long long)(b * 6 + h) * Lqp + q) * 128 + d];
}

// kv = concat(enc[B*222,512], dec[B*168,512]) -> [B*390,512] f16
__global__ void k_ckv(const _Float16* __restrict__ X, const _Float16* __restrict__ D,
                      _Float16* __restrict__ KV, long long n)
{
    long long i = (long long)blockIdx.x * 256 + threadIdx.x;
    if (i >= n) return;
    int c = (int)(i % 512); long long r = i / 512;
    int l = (int)(r % SKV); int b = (int)(r / SKV);
    KV[i] = (l < SEQE) ? X[((long long)b * SEQE + l) * 512 + c]
                       : D[((long long)b * LDQ + (l - SEQE)) * 512 + c];
}

// final projection: out[row,c] = dot(dec[row,:512], pred_w[:,c]),  c<7
__global__ void k_pred(const float* __restrict__ X, const float* __restrict__ W,
                       float* __restrict__ O, long long n)
{
    long long i = (long long)blockIdx.x * 256 + threadIdx.x;
    if (i >= n) return;
    int c = (int)(i % 7); long long row = i / 7;
    const float* xr = X + row * 512;
    float s = 0.f;
    for (int k = 0; k < 512; ++k) s += xr[k] * W[k * 7 + c];
    O[i] = s;
}

// ===========================================================================
extern "C" void kernel_launch(void* const* d_in, const int* in_sizes, int n_in,
                              void* d_out, int out_size, void* d_ws, size_t ws_size,
                              hipStream_t stream)
{
    (void)in_sizes; (void)n_in; (void)out_size; (void)ws_size;

    const float* x_enc      = (const float*)d_in[0];
    const float* t_enc      = (const float*)d_in[1];
    const float* x_dec      = (const float*)d_in[2];
    const float* t_dec      = (const float*)d_in[3];
    const float* enc_tok_w  = (const float*)d_in[4];
    const float* enc_time_w = (const float*)d_in[5];
    const float* enc_time_b = (const float*)d_in[6];
    const float* dec_tok_w  = (const float*)d_in[7];
    const float* dec_time_w = (const float*)d_in[8];
    const float* dec_time_b = (const float*)d_in[9];
    const float* bc_down_w  = (const float*)d_in[10];
    const float* bc_down_b  = (const float*)d_in[11];
    const float* bc_conv_w  = (const float*)d_in[12];
    const float* bc_conv_b  = (const float*)d_in[13];
    const float* bc_bn_g    = (const float*)d_in[14];
    const float* bc_bn_b    = (const float*)d_in[15];
    const float* bc_bn_m    = (const float*)d_in[16];
    const float* bc_bn_v    = (const float*)d_in[17];
    const float* bc_up_w    = (const float*)d_in[18];
    const float* bc_up_b    = (const float*)d_in[19];
    const float* bc_ln_g    = (const float*)d_in[20];
    const float* bc_ln_b    = (const float*)d_in[21];
    const float* enc_wq = (const float*)d_in[22]; const float* enc_bq = (const float*)d_in[23];
    const float* enc_wk = (const float*)d_in[24]; const float* enc_bk = (const float*)d_in[25];
    const float* enc_wv = (const float*)d_in[26]; const float* enc_bv = (const float*)d_in[27];
    const float* enc_wo = (const float*)d_in[28]; const float* enc_bo = (const float*)d_in[29];
    const float* enc_ln1_g = (const float*)d_in[30]; const float* enc_ln1_b = (const float*)d_in[31];
    const float* enc_w1 = (const float*)d_in[32]; const float* enc_b1 = (const float*)d_in[33];
    const float* enc_w2 = (const float*)d_in[34]; const float* enc_b2 = (const float*)d_in[35];
    const float* enc_ln2_g = (const float*)d_in[36]; const float* enc_ln2_b = (const float*)d_in[37];
    const float* dec_wq = (const float*)d_in[38]; const float* dec_bq = (const float*)d_in[39];
    const float* dec_wk = (const float*)d_in[40]; const float* dec_bk = (const float*)d_in[41];
    const float* dec_wv = (const float*)d_in[42]; const float* dec_bv = (const float*)d_in[43];
    const float* dec_wo = (const float*)d_in[44]; const float* dec_bo = (const float*)d_in[45];
    const float* dec_ln1_g = (const float*)d_in[46]; const float* dec_ln1_b = (const float*)d_in[47];
    const float* dec_w1 = (const float*)d_in[48]; const float* dec_b1 = (const float*)d_in[49];
    const float* dec_w2 = (const float*)d_in[50]; const float* dec_b2 = (const float*)d_in[51];
    const float* dec_ln2_g = (const float*)d_in[52]; const float* dec_ln2_b = (const float*)d_in[53];
    const float* pred_w = (const float*)d_in[54];
    float* out = (float*)d_out;

    // --- workspace bump allocator ------------------------------------------
    char* wsb = (char*)d_ws;
    size_t off = 0;
    auto alloc = [&](size_t bytes) -> void* {
        void* p = wsb + off;
        off = (off + bytes + 255) & ~(size_t)255;
        return p;
    };

    _Float16* zbuf   = (_Float16*)alloc((size_t)ZBUF_HALFS * 2);
    _Float16* encQT  = (_Float16*)alloc((size_t)6 * 768 * 512 * 2);
    _Float16* encKT  = (_Float16*)alloc((size_t)6 * 768 * 512 * 2);
    _Float16* encVT  = (_Float16*)alloc((size_t)6 * 768 * 512 * 2);
    _Float16* encOT  = (_Float16*)alloc((size_t)6 * 512 * 768 * 2);
    _Float16* encW1T = (_Float16*)alloc((size_t)6 * 512 * 512 * 2);
    _Float16* encW2T = (_Float16*)alloc((size_t)6 * 512 * 512 * 2);
    _Float16* decQT  = (_Float16*)alloc((size_t)2 * 768 * 512 * 2);
    _Float16* decKT  = (_Float16*)alloc((size_t)2 * 768 * 512 * 2);
    _Float16* decVT  = (_Float16*)alloc((size_t)2 * 768 * 512 * 2);
    _Float16* decOT  = (_Float16*)alloc((size_t)2 * 512 * 768 * 2);
    _Float16* decW1T = (_Float16*)alloc((size_t)2 * 512 * 512 * 2);
    _Float16* decW2T = (_Float16*)alloc((size_t)2 * 512 * 512 * 2);
    _Float16* downT  = (_Float16*)alloc((size_t)128 * 512 * 2);
    _Float16* upT    = (_Float16*)alloc((size_t)512 * 128 * 2);

    float*    E32   = (float*)   alloc((size_t)NB * LE * 512 * 4);
    _Float16* E16   = (_Float16*)alloc((size_t)NB * LE * 512 * 2);
    float*    Dst32 = (float*)   alloc((size_t)NB * LDQ * 512 * 4);
    _Float16* Dst16 = (_Float16*)alloc((size_t)NB * LDQ * 512 * 2);
    float*    down32= (float*)   alloc((size_t)NB * LE * 128 * 4);
    float*    cat32 = (float*)   alloc((size_t)NB * 54 * 128 * 4);
    _Float16* cat16 = (_Float16*)alloc((size_t)NB * 54 * 128 * 2);
    float*    up32  = (float*)   alloc((size_t)NB * 54 * 512 * 4);
    float*    X32   = (float*)   alloc((size_t)NB * SEQE * 512 * 4);
    _Float16* X16   = (_Float16*)alloc((size_t)NB * SEQE * 512 * 2);
    _Float16* KV16  = (_Float16*)alloc((size_t)NB * SKV * 512 * 2);
    float*    Q32   = (float*)   alloc((size_t)NB * SEQE * 768 * 4);  // queries (max B*222 rows)
    _Float16* Q16   = (_Float16*)alloc((size_t)NB * SEQE * 768 * 2);
    float*    K32   = (float*)   alloc((size_t)NB * SKV * 768 * 4);   // keys (max B*390 rows)
    _Float16* K16   = (_Float16*)alloc((size_t)NB * SKV * 768 * 2);
    float*    V32   = (float*)   alloc((size_t)NB * SKV * 768 * 4);
    _Float16* Vt16  = (_Float16*)alloc((size_t)NB * 6 * 128 * NKD * 2);
    float*    S32   = (float*)   alloc((size_t)NB * 6 * MQD * NKD * 4); // covers enc 224x256 too
    _Float16* S16   = (_Float16*)alloc((size_t)NB * 6 * MQD * NKD * 2);
    float*    AV32  = (float*)   alloc((size_t)NB * 6 * MQE * 128 * 4);
    _Float16* M16   = (_Float16*)alloc((size_t)NB * SEQE * 768 * 2);
    float*    P32   = (float*)   alloc((size_t)NB * SEQE * 512 * 4);
    float*    H32   = Q32;   // FFN hidden aliases Q (disjoint lifetime)
    _Float16* H16   = Q16;

    const float scale = 0.08838834764831845f;  // 1/sqrt(128)

    auto gemm = [&](const _Float16* A, int lda, long long sAb, long long sAh, int validM,
                    const _Float16* Bt, int ldb, long long sBb, long long sBh, int validN,
                    float* C, int ldc, long long sCb, long long sCh,
                    const float* bias, int M, int N, int K, int batches, int nh) {
        dim3 g(N / 128, M / 16, batches);
        k_wmma_gemm<<<g, 128, 0, stream>>>(A, lda, sAb, sAh, validM,
                                           Bt, ldb, sBb, sBh, validN,
                                           C, ldc, sCb, sCh, bias, K, nh, zbuf);
    };
    auto wt = [&](const float* W, _Float16* Wt, int K, int N) {
        long long n = (long long)K * N;
        k_wT<<<(unsigned)((n + 255) / 256), 256, 0, stream>>>(W, Wt, K, N);
    };
    auto tohalf = [&](const float* s, _Float16* d, long long n) {
        k_h<<<(unsigned)((n + 255) / 256), 256, 0, stream>>>(s, d, n);
    };

    // --- Phase 0: zero scratch + transpose-convert weights to f16 ----------
    k_zero16<<<(ZBUF_HALFS + 255) / 256, 256, 0, stream>>>(zbuf, ZBUF_HALFS);
    for (int i = 0; i < 6; ++i) {
        wt(enc_wq + (size_t)i * 512 * 768, encQT + (size_t)i * 768 * 512, 512, 768);
        wt(enc_wk + (size_t)i * 512 * 768, encKT + (size_t)i * 768 * 512, 512, 768);
        wt(enc_wv + (size_t)i * 512 * 768, encVT + (size_t)i * 768 * 512, 512, 768);
        wt(enc_wo + (size_t)i * 768 * 512, encOT + (size_t)i * 512 * 768, 768, 512);
        wt(enc_w1 + (size_t)i * 512 * 512, encW1T + (size_t)i * 512 * 512, 512, 512);
        wt(enc_w2 + (size_t)i * 512 * 512, encW2T + (size_t)i * 512 * 512, 512, 512);
    }
    for (int i = 0; i < 2; ++i) {
        wt(dec_wq + (size_t)i * 512 * 768, decQT + (size_t)i * 768 * 512, 512, 768);
        wt(dec_wk + (size_t)i * 512 * 768, decKT + (size_t)i * 768 * 512, 512, 768);
        wt(dec_wv + (size_t)i * 512 * 768, decVT + (size_t)i * 768 * 512, 512, 768);
        wt(dec_wo + (size_t)i * 768 * 512, decOT + (size_t)i * 512 * 768, 768, 512);
        wt(dec_w1 + (size_t)i * 512 * 512, decW1T + (size_t)i * 512 * 512, 512, 512);
        wt(dec_w2 + (size_t)i * 512 * 512, decW2T + (size_t)i * 512 * 512, 512, 512);
    }
    wt(bc_down_w, downT, 512, 128);
    wt(bc_up_w,   upT,   128, 512);

    // --- Phase 1: embeddings ------------------------------------------------
    k_embed<<<NB * LE, 512, 0, stream>>>(x_enc, t_enc, enc_tok_w, enc_time_w, enc_time_b, E32, E16, LE);
    k_embed<<<NB * LDQ, 512, 0, stream>>>(x_dec, t_dec, dec_tok_w, dec_time_w, dec_time_b, Dst32, Dst16, LDQ);

    // --- Phase 2: bottleneck ------------------------------------------------
    gemm(E16, 512, 0, 0, NB * LE, downT, 512, 0, 0, 128,
         down32, 128, 0, 0, bc_down_b, NB * LE, 128, 512, 1, 1);
    k_conv_bn_elu<<<NB * 42, 128, 0, stream>>>(down32, (long long)LE * 128,
        bc_conv_w, bc_conv_b, bc_bn_g, bc_bn_b, bc_bn_m, bc_bn_v,
        cat32, (long long)54 * 128, 0, 42);
    k_conv_bn_elu<<<NB * 10, 128, 0, stream>>>(cat32, (long long)54 * 128,
        bc_conv_w + 128 * 128 * 4, bc_conv_b + 128, bc_bn_g + 128, bc_bn_b + 128,
        bc_bn_m + 128, bc_bn_v + 128, cat32, (long long)54 * 128, 42, 10);
    k_conv_bn_elu<<<NB * 2, 128, 0, stream>>>(cat32 + 42 * 128, (long long)54 * 128,
        bc_conv_w + 2 * 128 * 128 * 4, bc_conv_b + 256, bc_bn_g + 256, bc_bn_b + 256,
        bc_bn_m + 256, bc_bn_v + 256, cat32, (long long)54 * 128, 52, 2);
    tohalf(cat32, cat16, (long long)NB * 54 * 128);
    gemm(cat16, 128, 0, 0, NB * 54, upT, 128, 0, 0, 512,
         up32, 512, 0, 0, bc_up_b, NB * 54, 512, 128, 1, 1);
    k_concat_ln<<<NB * SEQE, 256, 0, stream>>>(E32, up32, bc_ln_g, bc_ln_b, X32, X16);

    // --- Phase 3: encoder layers -------------------------------------------
    for (int i = 0; i < 6; ++i) {
        const _Float16* QT = encQT + (size_t)i * 768 * 512;
        const _Float16* KT = encKT + (size_t)i * 768 * 512;
        const _Float16* VT = encVT + (size_t)i * 768 * 512;
        const _Float16* OT = encOT + (size_t)i * 512 * 768;
        const _Float16* W1 = encW1T + (size_t)i * 512 * 512;
        const _Float16* W2 = encW2T + (size_t)i * 512 * 512;

        gemm(X16, 512, 0, 0, NB * SEQE, QT, 512, 0, 0, 768,
             Q32, 768, 0, 0, enc_bq + i * 768, NB * SEQE, 768, 512, 1, 1);
        tohalf(Q32, Q16, (long long)NB * SEQE * 768);
        gemm(X16, 512, 0, 0, NB * SEQE, KT, 512, 0, 0, 768,
             K32, 768, 0, 0, enc_bk + i * 768, NB * SEQE, 768, 512, 1, 1);
        tohalf(K32, K16, (long long)NB * SEQE * 768);
        gemm(X16, 512, 0, 0, NB * SEQE, VT, 512, 0, 0, 768,
             V32, 768, 0, 0, enc_bv + i * 768, NB * SEQE, 768, 512, 1, 1);
        {
            long long n = (long long)NB * 6 * 128 * NKE;
            k_vt<<<(unsigned)((n + 255) / 256), 256, 0, stream>>>(V32, Vt16, SEQE, NKE, n);
        }
        // scores: per (b,h), [224x256] = Q[222x128] * K^T
        gemm(Q16, 768, (long long)SEQE * 768, 128, SEQE,
             K16, 768, (long long)SEQE * 768, 128, SEQE,
             S32, NKE, (long long)6 * MQE * NKE, (long long)MQE * NKE,
             nullptr, MQE, NKE, 128, NB * 6, 6);
        k_softmax<<<dim3(MQE, NB * 6), 256, 0, stream>>>(S32, S16, SEQE, MQE, SEQE, NKE, scale, 0);
        // AV: per (b,h), [224x128] = P[224x256] * V[256x128]
        gemm(S16, NKE, (long long)6 * MQE * NKE, (long long)MQE * NKE, MQE,
             Vt16, NKE, (long long)6 * 128 * NKE, (long long)128 * NKE, 128,
             AV32, 128, (long long)6 * MQE * 128, (long long)MQE * 128,
             nullptr, MQE, 128, NKE, NB * 6, 6);
        {
            long long n = (long long)NB * SEQE * 768;
            k_merge<<<(unsigned)((n + 255) / 256), 256, 0, stream>>>(AV32, M16, SEQE, MQE, n);
        }
        gemm(M16, 768, 0, 0, NB * SEQE, OT, 768, 0, 0, 512,
             P32, 512, 0, 0, enc_bo + i * 512, NB * SEQE, 512, 768, 1, 1);
        k_ln_res<<<NB * SEQE, 256, 0, stream>>>(P32, X32, enc_ln1_g + i * 512, enc_ln1_b + i * 512,
                                                X32, X16, 1e-6f);
        gemm(X16, 512, 0, 0, NB * SEQE, W1, 512, 0, 0, 512,
             H32, 512, 0, 0, enc_b1 + i * 512, NB * SEQE, 512, 512, 1, 1);
        {
            long long n = (long long)NB * SEQE * 512;
            k_gelu<<<(unsigned)((n + 255) / 256), 256, 0, stream>>>(H32, H16, n);
        }
        gemm(H16, 512, 0, 0, NB * SEQE, W2, 512, 0, 0, 512,
             P32, 512, 0, 0, enc_b2 + i * 512, NB * SEQE, 512, 512, 1, 1);
        k_ln_res<<<NB * SEQE, 256, 0, stream>>>(P32, X32, enc_ln2_g + i * 512, enc_ln2_b + i * 512,
                                                X32, X16, 1e-6f);
    }

    // --- Phase 4: decoder layers -------------------------------------------
    for (int i = 0; i < 2; ++i) {
        const _Float16* QT = decQT + (size_t)i * 768 * 512;
        const _Float16* KT = decKT + (size_t)i * 768 * 512;
        const _Float16* VT = decVT + (size_t)i * 768 * 512;
        const _Float16* OT = decOT + (size_t)i * 512 * 768;
        const _Float16* W1 = decW1T + (size_t)i * 512 * 512;
        const _Float16* W2 = decW2T + (size_t)i * 512 * 512;

        {
            long long n = (long long)NB * SKV * 512;
            k_ckv<<<(unsigned)((n + 255) / 256), 256, 0, stream>>>(X16, Dst16, KV16, n);
        }
        gemm(Dst16, 512, 0, 0, NB * LDQ, QT, 512, 0, 0, 768,
             Q32, 768, 0, 0, dec_bq + i * 768, NB * LDQ, 768, 512, 1, 1);
        tohalf(Q32, Q16, (long long)NB * LDQ * 768);
        gemm(KV16, 512, 0, 0, NB * SKV, KT, 512, 0, 0, 768,
             K32, 768, 0, 0, dec_bk + i * 768, NB * SKV, 768, 512, 1, 1);
        tohalf(K32, K16, (long long)NB * SKV * 768);
        gemm(KV16, 512, 0, 0, NB * SKV, VT, 512, 0, 0, 768,
             V32, 768, 0, 0, dec_bv + i * 768, NB * SKV, 768, 512, 1, 1);
        {
            long long n = (long long)NB * 6 * 128 * NKD;
            k_vt<<<(unsigned)((n + 255) / 256), 256, 0, stream>>>(V32, Vt16, SKV, NKD, n);
        }
        gemm(Q16, 768, (long long)LDQ * 768, 128, LDQ,
             K16, 768, (long long)SKV * 768, 128, SKV,
             S32, NKD, (long long)6 * MQD * NKD, (long long)MQD * NKD,
             nullptr, MQD, NKD, 128, NB * 6, 6);
        k_softmax<<<dim3(MQD, NB * 6), 256, 0, stream>>>(S32, S16, LDQ, MQD, SKV, NKD, scale, 1);
        gemm(S16, NKD, (long long)6 * MQD * NKD, (long long)MQD * NKD, MQD,
             Vt16, NKD, (long long)6 * 128 * NKD, (long long)128 * NKD, 128,
             AV32, 128, (long long)6 * MQD * 128, (long long)MQD * 128,
             nullptr, MQD, 128, NKD, NB * 6, 6);
        {
            long long n = (long long)NB * LDQ * 768;
            k_merge<<<(unsigned)((n + 255) / 256), 256, 0, stream>>>(AV32, M16, LDQ, MQD, n);
        }
        gemm(M16, 768, 0, 0, NB * LDQ, OT, 768, 0, 0, 512,
             P32, 512, 0, 0, dec_bo + i * 512, NB * LDQ, 512, 768, 1, 1);
        k_ln_res<<<NB * LDQ, 256, 0, stream>>>(P32, Dst32, dec_ln1_g + i * 512, dec_ln1_b + i * 512,
                                               Dst32, Dst16, 1e-6f);
        gemm(Dst16, 512, 0, 0, NB * LDQ, W1, 512, 0, 0, 512,
             H32, 512, 0, 0, dec_b1 + i * 512, NB * LDQ, 512, 512, 1, 1);
        {
            long long n = (long long)NB * LDQ * 512;
            k_gelu<<<(unsigned)((n + 255) / 256), 256, 0, stream>>>(H32, H16, n);
        }
        gemm(H16, 512, 0, 0, NB * LDQ, W2, 512, 0, 0, 512,
             P32, 512, 0, 0, dec_b2 + i * 512, NB * LDQ, 512, 512, 1, 1);
        k_ln_res<<<NB * LDQ, 256, 0, stream>>>(P32, Dst32, dec_ln2_g + i * 512, dec_ln2_b + i * 512,
                                               Dst32, Dst16, 1e-6f);
    }

    // --- Phase 5: prediction head ------------------------------------------
    {
        long long n = (long long)NB * LDQ * 7;
        k_pred<<<(unsigned)((n + 255) / 256), 256, 0, stream>>>(Dst32, pred_w, out, n);
    }
}
